// Attention_88656714925156
// MI455X (gfx1250) — matrix operand order repaired and checked by reading, not compiled, not run on hardware
//
#include <hip/hip_runtime.h>
#include <cstdint>

// ---------------------------------------------------------------------------
// Types for CDNA5 WMMA (gfx1250, wave32)
// ---------------------------------------------------------------------------
typedef __attribute__((ext_vector_type(16))) __bf16 v16bf;
typedef __attribute__((ext_vector_type(8)))  float  v8f;

union FragB {            // 16 bf16 = 32 bytes = two 128-bit loads
    uint4 q[2];
    v16bf v;
};

__device__ __forceinline__ uint16_t f2bf(float f) {
    uint32_t u = __float_as_uint(f);
    u += 0x7fffu + ((u >> 16) & 1u);   // round-to-nearest-even
    return (uint16_t)(u >> 16);
}

// ---------------------------------------------------------------------------
// CDNA5 async memory->LDS DMA (ASYNCcnt-tracked, no VGPR round-trip).
// Generic LDS pointers carry the LDS byte offset in the low 32 bits.
// ---------------------------------------------------------------------------
__device__ __forceinline__ void async_ld_b128(void* lds_ptr, const void* gptr) {
    uint32_t lds = (uint32_t)(uintptr_t)lds_ptr;
    uint64_t ga  = (uint64_t)(uintptr_t)gptr;
    asm volatile("global_load_async_to_lds_b128 %0, %1, off"
                 :: "v"(lds), "v"(ga) : "memory");
}
__device__ __forceinline__ void wait_async0() {
    asm volatile("s_wait_asynccnt 0" ::: "memory");
}

// ---------------------------------------------------------------------------
// K0: fp32 -> bf16 conversion (vectorized 4/thread)
// ---------------------------------------------------------------------------
__global__ __launch_bounds__(256) void k_cvt(const float* __restrict__ src,
                                             uint16_t* __restrict__ dst, int n) {
    int i = (blockIdx.x * blockDim.x + threadIdx.x) * 4;
    if (i < n) {
        float4 f = *(const float4*)(src + i);
        uint2 o;
        o.x = (uint32_t)f2bf(f.x) | ((uint32_t)f2bf(f.y) << 16);
        o.y = (uint32_t)f2bf(f.z) | ((uint32_t)f2bf(f.w) << 16);
        *(uint2*)(dst + i) = o;
    }
}

// ---------------------------------------------------------------------------
// Shared GEMM machinery: 256 thr = 8 waves, tile 128(M) x 64(N), K-step 32,
// double-buffered LDS fed by async DMA.
// ---------------------------------------------------------------------------
#define LDSTRIDE 40   // 32 + 8 pad (keeps 16B alignment, breaks bank conflicts)

__device__ __forceinline__ void stage_tile_async(
        const uint16_t* __restrict__ A, const uint16_t* __restrict__ W,
        uint16_t* AsBuf, uint16_t* WsBuf, int bm, int bn, int k0, int tid) {
#pragma unroll
    for (int i = 0; i < 2; ++i) {              // A tile: 128 x 32
        int c = tid + i * 256;
        int row = c >> 2, col = (c & 3) * 8;
        async_ld_b128(&AsBuf[row * LDSTRIDE + col],
                      A + (size_t)(bm + row) * 1024 + k0 + col);
    }
    {                                           // W tile: 64 x 32
        int row = tid >> 2, col = (tid & 3) * 8;
        async_ld_b128(&WsBuf[row * LDSTRIDE + col],
                      W + (size_t)(bn + row) * 1024 + k0 + col);
    }
}

// ---------------------------------------------------------------------------
// K1: QKV projection GEMM.  C[4096][3072] = Xb[4096][1024] * Wqkv[3072][1024]^T
// Epilogue scatters into Q [bh][n][dh] (scaled), K [bh][n][dh], Vt [bh][dh][n]
// ---------------------------------------------------------------------------
__global__ __launch_bounds__(256) void k_gemm_qkv(
        const uint16_t* __restrict__ A,   // [4096][1024] bf16
        const uint16_t* __restrict__ W,   // [3072][1024] bf16
        uint16_t* __restrict__ Qo, uint16_t* __restrict__ Ko,
        uint16_t* __restrict__ Vt) {
    __shared__ uint16_t As[2][128 * LDSTRIDE];
    __shared__ uint16_t Ws[2][64 * LDSTRIDE];

    const int tid  = threadIdx.x;
    const int lane = tid & 31;
    const int wave = tid >> 5;
    const int wm   = wave & 3;          // 4 waves in M
    const int wn   = wave >> 2;         // 2 waves in N
    const int ln15 = lane & 15;
    const int hi   = lane >> 4;
    const int bm   = blockIdx.y * 128;
    const int bn   = blockIdx.x * 64;

    v8f acc[2][2] = {};

    stage_tile_async(A, W, As[0], Ws[0], bm, bn, 0, tid);
    wait_async0();
    __syncthreads();

    for (int it = 0; it < 32; ++it) {
        const int cur = it & 1;
        if (it + 1 < 32)                // DMA next tile while computing this one
            stage_tile_async(A, W, As[cur ^ 1], Ws[cur ^ 1], bm, bn, (it + 1) * 32, tid);

        // ---- fragment loads (ISA VGPR layouts) ----
        FragB af[2], wf[2];
#pragma unroll
        for (int i = 0; i < 2; ++i) {            // A: lanes0-15 K{0-7,16-23}
            int m  = wm * 32 + i * 16 + ln15;
            int lo = hi * 8;
            af[i].q[0] = *(const uint4*)(&As[cur][m * LDSTRIDE + lo]);
            af[i].q[1] = *(const uint4*)(&As[cur][m * LDSTRIDE + lo + 16]);
        }
#pragma unroll
        for (int j = 0; j < 2; ++j) {            // B: lanes0-15 K{0-7,16-23}
            int n  = wn * 32 + j * 16 + ln15;
            int kb = hi * 8;
            wf[j].q[0] = *(const uint4*)(&Ws[cur][n * LDSTRIDE + kb]);
            wf[j].q[1] = *(const uint4*)(&Ws[cur][n * LDSTRIDE + kb + 16]);
        }
#pragma unroll
        for (int i = 0; i < 2; ++i)
#pragma unroll
            for (int j = 0; j < 2; ++j)
                acc[i][j] = __builtin_amdgcn_wmma_f32_16x16x32_bf16(
                    false, af[i].v, false, wf[j].v, (short)0, acc[i][j], false, false);

        wait_async0();                  // next tile landed
        __syncthreads();                // everyone done reading cur buffer
    }

    // ---- epilogue: scatter to Q (scaled) / K / V-transposed, bf16 ----
    const int ntype = bn >> 10;                 // 0=Q 1=K 2=V
    const int h     = (bn & 1023) >> 6;
#pragma unroll
    for (int i = 0; i < 2; ++i)
#pragma unroll
        for (int j = 0; j < 2; ++j)
#pragma unroll
            for (int r = 0; r < 8; ++r) {
                int m  = bm + wm * 32 + i * 16 + r + hi * 8;
                int n  = bn + wn * 32 + j * 16 + ln15;
                int b  = m >> 11, nq = m & 2047;
                int dh = n & 63;
                int bh = b * 16 + h;
                float v = acc[i][j][r];
                if (ntype == 0)
                    Qo[((size_t)bh * 2048 + nq) * 64 + dh] = f2bf(v * 0.125f); // fold DH^-0.5
                else if (ntype == 1)
                    Ko[((size_t)bh * 2048 + nq) * 64 + dh] = f2bf(v);
                else
                    Vt[((size_t)bh * 64 + dh) * 2048 + nq] = f2bf(v);
            }
}

// ---------------------------------------------------------------------------
// K2: causal flash attention.  Block = (b,h) x 64 q-rows, 4 waves x 16 rows.
// S = Q K^T via 2x(16x16x32) WMMA per 16-key subtile, online softmax in f32,
// O += P V via WMMA with P round-tripped through wave-private LDS.
// K/V fragments read directly from L2-resident global (fits in 192MB L2).
// ---------------------------------------------------------------------------
__global__ __launch_bounds__(128) void k_flash(
        const uint16_t* __restrict__ Q,   // [32][2048][64] bf16 (pre-scaled)
        const uint16_t* __restrict__ K,   // [32][2048][64] bf16
        const uint16_t* __restrict__ Vt,  // [32][64][2048]  bf16 (d-major)
        uint16_t* __restrict__ AO) {      // [4096][1024]   bf16
    const int bh   = blockIdx.y;
    const int q0   = blockIdx.x * 64;
    const int tid  = threadIdx.x;
    const int lane = tid & 31;
    const int wave = tid >> 5;
    const int ln15 = lane & 15;
    const int hi   = lane >> 4;
    const int qw   = q0 + wave * 16;          // this wave's first q row

    __shared__ uint16_t Plds[4][16 * LDSTRIDE];   // per-wave P scratch (16x32)

    const uint16_t* Qb = Q  + (size_t)bh * 2048 * 64;
    const uint16_t* Kb = K  + (size_t)bh * 2048 * 64;
    const uint16_t* Vb = Vt + (size_t)bh * 64 * 2048;

    // Q A-fragments for both K-dim halves (d 0..31, 32..63), resident all loop
    FragB qf[2];
    {
        int q  = qw + ln15;
        int lo = hi * 8;
        qf[0].q[0] = *(const uint4*)(Qb + (size_t)q * 64 + lo);
        qf[0].q[1] = *(const uint4*)(Qb + (size_t)q * 64 + lo + 16);
        qf[1].q[0] = *(const uint4*)(Qb + (size_t)q * 64 + 32 + lo);
        qf[1].q[1] = *(const uint4*)(Qb + (size_t)q * 64 + 32 + lo + 16);
    }

    v8f   o[4] = {};                         // 4 d-tiles of 16 cols, f32 accum
    float mrow[8], lrow[8];
#pragma unroll
    for (int r = 0; r < 8; ++r) { mrow[r] = -INFINITY; lrow[r] = 0.f; }

    const int kend = qw + 16;                // causal: keys < qw+16 needed
    for (int k0 = 0; k0 < kend; k0 += 32) {
        // prefetch next K-step fragments toward the WGP while we compute
        if (k0 + 32 < kend) {
            __builtin_prefetch(Kb + (size_t)(k0 + 32 + ln15) * 64 + hi * 16, 0, 3);
            __builtin_prefetch(Vb + (size_t)ln15 * 2048 + k0 + 32 + hi * 16, 0, 3);
        }

        // ---- S = Q K^T for 32 keys (two 16-col C tiles) ----
        v8f s[2] = {};
#pragma unroll
        for (int t = 0; t < 2; ++t) {
            int kc = k0 + t * 16;
#pragma unroll
            for (int half = 0; half < 2; ++half) {
                FragB kf;
                int n  = kc + ln15;
                int kb = half * 32 + hi * 8;
                kf.q[0] = *(const uint4*)(Kb + (size_t)n * 64 + kb);
                kf.q[1] = *(const uint4*)(Kb + (size_t)n * 64 + kb + 16);
                s[t] = __builtin_amdgcn_wmma_f32_16x16x32_bf16(
                    false, qf[half].v, false, kf.v, (short)0, s[t], false, false);
            }
        }

        // ---- causal mask + online softmax (rows: r + hi*8, cols: ln15) ----
        float alpha[8];
#pragma unroll
        for (int r = 0; r < 8; ++r) {
            int qg = qw + r + hi * 8;
#pragma unroll
            for (int t = 0; t < 2; ++t) {
                int kg = k0 + t * 16 + ln15;
                if (kg > qg) s[t][r] = -INFINITY;
            }
            float mx = fmaxf(s[0][r], s[1][r]);
#pragma unroll
            for (int d = 1; d < 16; d <<= 1) mx = fmaxf(mx, __shfl_xor(mx, d, 32));
            float mnew = fmaxf(mrow[r], mx);
            alpha[r] = __expf(mrow[r] - mnew);
            float rs = 0.f;
#pragma unroll
            for (int t = 0; t < 2; ++t) {
                float p = __expf(s[t][r] - mnew);
                s[t][r] = p;
                rs += p;
            }
#pragma unroll
            for (int d = 1; d < 16; d <<= 1) rs += __shfl_xor(rs, d, 32);
            lrow[r] = lrow[r] * alpha[r] + rs;
            mrow[r] = mnew;
        }

        // ---- P: C-layout -> LDS -> A-layout fragment (wave-private) ----
        uint16_t* P = &Plds[wave][0];
#pragma unroll
        for (int t = 0; t < 2; ++t)
#pragma unroll
            for (int r = 0; r < 8; ++r)
                P[(r + hi * 8) * LDSTRIDE + t * 16 + ln15] = f2bf(s[t][r]);
        FragB pf;
        {
            int lo = hi * 8;
            pf.q[0] = *(const uint4*)(&P[ln15 * LDSTRIDE + lo]);
            pf.q[1] = *(const uint4*)(&P[ln15 * LDSTRIDE + lo + 16]);
        }

        // ---- O = O*alpha + P @ V (4 output d-tiles) ----
#pragma unroll
        for (int jd = 0; jd < 4; ++jd) {
#pragma unroll
            for (int r = 0; r < 8; ++r) o[jd][r] *= alpha[r];
            FragB vf;
            int d  = jd * 16 + ln15;
            int kb = hi * 8;
            vf.q[0] = *(const uint4*)(Vb + (size_t)d * 2048 + k0 + kb);
            vf.q[1] = *(const uint4*)(Vb + (size_t)d * 2048 + k0 + kb + 16);
            o[jd] = __builtin_amdgcn_wmma_f32_16x16x32_bf16(
                false, pf.v, false, vf.v, (short)0, o[jd], false, false);
        }
    }

    // ---- finalize: O /= l, store bf16 into [B*N][D] at col h*64+d ----
    const int b = bh >> 4, h = bh & 15;
    float inv_l[8];
#pragma unroll
    for (int r = 0; r < 8; ++r) inv_l[r] = 1.f / lrow[r];
#pragma unroll
    for (int jd = 0; jd < 4; ++jd)
#pragma unroll
        for (int r = 0; r < 8; ++r) {
            int m = b * 2048 + qw + r + hi * 8;
            int c = h * 64 + jd * 16 + ln15;
            AO[(size_t)m * 1024 + c] = f2bf(o[jd][r] * inv_l[r]);
        }
}

// ---------------------------------------------------------------------------
// K3: output projection. out[4096][1024] = AO[4096][1024] * Wout[1024][1024]^T
// Same async double-buffered tiling as K1; fp32 epilogue straight to d_out.
// ---------------------------------------------------------------------------
__global__ __launch_bounds__(256) void k_gemm_out(
        const uint16_t* __restrict__ A,   // [4096][1024] bf16
        const uint16_t* __restrict__ W,   // [1024][1024] bf16
        float* __restrict__ Out) {
    __shared__ uint16_t As[2][128 * LDSTRIDE];
    __shared__ uint16_t Ws[2][64 * LDSTRIDE];

    const int tid  = threadIdx.x;
    const int lane = tid & 31;
    const int wave = tid >> 5;
    const int wm   = wave & 3;
    const int wn   = wave >> 2;
    const int ln15 = lane & 15;
    const int hi   = lane >> 4;
    const int bm   = blockIdx.y * 128;
    const int bn   = blockIdx.x * 64;

    v8f acc[2][2] = {};

    stage_tile_async(A, W, As[0], Ws[0], bm, bn, 0, tid);
    wait_async0();
    __syncthreads();

    for (int it = 0; it < 32; ++it) {
        const int cur = it & 1;
        if (it + 1 < 32)
            stage_tile_async(A, W, As[cur ^ 1], Ws[cur ^ 1], bm, bn, (it + 1) * 32, tid);

        FragB af[2], wf[2];
#pragma unroll
        for (int i = 0; i < 2; ++i) {
            int m  = wm * 32 + i * 16 + ln15;
            int lo = hi * 8;
            af[i].q[0] = *(const uint4*)(&As[cur][m * LDSTRIDE + lo]);
            af[i].q[1] = *(const uint4*)(&As[cur][m * LDSTRIDE + lo + 16]);
        }
#pragma unroll
        for (int j = 0; j < 2; ++j) {
            int n  = wn * 32 + j * 16 + ln15;
            int kb = hi * 8;
            wf[j].q[0] = *(const uint4*)(&Ws[cur][n * LDSTRIDE + kb]);
            wf[j].q[1] = *(const uint4*)(&Ws[cur][n * LDSTRIDE + kb + 16]);
        }
#pragma unroll
        for (int i = 0; i < 2; ++i)
#pragma unroll
            for (int j = 0; j < 2; ++j)
                acc[i][j] = __builtin_amdgcn_wmma_f32_16x16x32_bf16(
                    false, af[i].v, false, wf[j].v, (short)0, acc[i][j], false, false);

        wait_async0();
        __syncthreads();
    }

#pragma unroll
    for (int i = 0; i < 2; ++i)
#pragma unroll
        for (int j = 0; j < 2; ++j)
#pragma unroll
            for (int r = 0; r < 8; ++r) {
                int m = bm + wm * 32 + i * 16 + r + hi * 8;
                int n = bn + wn * 32 + j * 16 + ln15;
                Out[(size_t)m * 1024 + n] = acc[i][j][r];
            }
}

// ---------------------------------------------------------------------------
// Host launcher.  Inputs: x [2,2048,1024] f32, mask (ignored; causal computed
// analytically), W_qkv [3072,1024] f32, W_out [1024,1024] f32.
// Workspace layout (uint16 elements):
//   xb 0 .. 4194304 | Wqkvb .. 7340032 | Woutb .. 8388608 |
//   Q .. 12582912 | K .. 16777216 | Vt .. 20971520 | AO .. 25165824  (48 MB)
// ---------------------------------------------------------------------------
extern "C" void kernel_launch(void* const* d_in, const int* in_sizes, int n_in,
                              void* d_out, int out_size, void* d_ws, size_t ws_size,
                              hipStream_t stream) {
    (void)in_sizes; (void)n_in; (void)out_size; (void)ws_size;
    const float* x    = (const float*)d_in[0];
    const float* Wqkv = (const float*)d_in[2];
    const float* Wout = (const float*)d_in[3];
    float* out = (float*)d_out;

    uint16_t* ws    = (uint16_t*)d_ws;
    uint16_t* xb    = ws;
    uint16_t* wqkvb = ws + 4194304;
    uint16_t* woutb = ws + 7340032;
    uint16_t* Qb    = ws + 8388608;
    uint16_t* Kb    = ws + 12582912;
    uint16_t* Vtb   = ws + 16777216;
    uint16_t* AOb   = ws + 20971520;

    k_cvt<<<4194304 / 1024, 256, 0, stream>>>(x,    xb,    4194304);
    k_cvt<<<3145728 / 1024, 256, 0, stream>>>(Wqkv, wqkvb, 3145728);
    k_cvt<<<1048576 / 1024, 256, 0, stream>>>(Wout, woutb, 1048576);

    k_gemm_qkv<<<dim3(48, 32), 256, 0, stream>>>(xb, wqkvb, Qb, Kb, Vtb);
    k_flash   <<<dim3(32, 32), 128, 0, stream>>>(Qb, Kb, Vtb, AOb);
    k_gemm_out<<<dim3(16, 32), 256, 0, stream>>>(AOb, woutb, out);
}
